// HiPPOMemPlainCell_31164282699978
// MI455X (gfx1250) — matrix operand order repaired and checked by reading, not compiled, not run on hardware
//
#include <hip/hip_runtime.h>
#include <math.h>

// ---------------- problem constants ----------------
#define BATCH   4
#define SEQLEN  128
#define HID     512
#define MSZ     256      // memory_dim
#define NORD    256      // memory_order
#define MN      (MSZ * NORD)      // 65536
#define KCHUNKS (MN / 32)         // 2048 WMMA K-chunks of 32

// ---------------- workspace layout (bytes) ----------------
// assumes ws_size >= ~137 MB (bf16 weight cache resident in 192MB L2)
#define OFF_H      0u                      // B*H fp32            (8192)
#define OFF_HID    8192u                   // B*H fp32            (8192)
#define OFF_G      16384u                  // B*H fp32            (8192)
#define OFF_M      24576u                  // B*N*M fp32 (m transposed [b][n][mi]) (1 MiB)
#define OFF_APACK  1073152u                // WMMA A-operand pack: K*16 bf16 (2 MiB)
#define OFF_WH     3170304u                // W_hxm bf16 (64 MiB)
#define OFF_WG     70279168u               // W_gxm bf16 (64 MiB)
#define ZERO_FLOATS 268288                 // h+hid+g+m region

typedef __attribute__((ext_vector_type(16))) __bf16 v16bf;
typedef __attribute__((ext_vector_type(8)))  float  v8f;

__device__ __forceinline__ unsigned short f2bf(float f) {
  unsigned u = __float_as_uint(f);
  unsigned r = u + 0x7FFFu + ((u >> 16) & 1u);   // round-to-nearest-even
  return (unsigned short)(r >> 16);
}

// ---------------- init: zero state ----------------
__global__ void k_zero(float* ws_f) {
  int i = blockIdx.x * blockDim.x + threadIdx.x;
  if (i < ZERO_FLOATS) ws_f[i] = 0.0f;
}

// ---------------- init: fp32 -> bf16 weight cache (L2-resident) ----------------
__global__ void k_cvt(const float* __restrict__ Wh, const float* __restrict__ Wg,
                      unsigned short* __restrict__ WhB, unsigned short* __restrict__ WgB) {
  size_t n = (size_t)HID * MN;
  for (size_t i = (size_t)blockIdx.x * blockDim.x + threadIdx.x; i < n;
       i += (size_t)gridDim.x * blockDim.x) {
    WhB[i] = f2bf(Wh[i]);
    WgB[i] = f2bf(Wg[i]);
  }
}

// ---------------- per-step memory update ----------------
// HiPPO-LegS A is lower triangular with A[n,k] = -r_n r_k (n>k), -(n+1) diag,
// r_n = sqrt(2n+1).  Both (A m) and the (I - dt/2 A)^{-1} forward substitution
// reduce to O(N) running prefix sums p, q per (batch, channel) thread.
// m stored transposed: m_t[b][n][mi]  (coalesced over mi = threadIdx.x)
__global__ void k_memstep(const float* __restrict__ ts,
                          const float* __restrict__ W_uxh,
                          const float* __restrict__ b_uxh,
                          const float* __restrict__ h,
                          float* __restrict__ m_t, int t) {
  int b  = blockIdx.x;        // 0..3
  int mi = threadIdx.x;       // 0..255
  float t1 = ts[b * SEQLEN + t];
  float tp = (t > 0) ? ts[b * SEQLEN + t - 1] : 0.0f;
  bool anyzero = false;
  #pragma unroll
  for (int bb = 0; bb < BATCH; ++bb) anyzero |= (ts[bb * SEQLEN + t] == 0.0f);
  float dt  = (t1 - tp) / ((t1 == 0.0f) ? 1.0f : t1);
  float hdt = 0.5f * dt;

  // u[b][mi] = dot(h[b,:], W_uxh[mi,:]) + b_uxh[mi]
  const float* hb = h + b * HID;
  const float* wr = W_uxh + (size_t)mi * HID;
  float u = b_uxh[mi];
  #pragma unroll 8
  for (int k = 0; k < HID; ++k) u = fmaf(hb[k], wr[k], u);

  float* mb = m_t + (size_t)b * MN;
  if (anyzero) {                       // reference: m := [u, 0, ..., 0]
    mb[mi] = u;                        // n == 0
    for (int n = 1; n < NORD; ++n) mb[n * MSZ + mi] = 0.0f;
    return;
  }
  float dtu = dt * u;
  float p = 0.0f, q = 0.0f;
  for (int n = 0; n < NORD; ++n) {
    float rn = __builtin_sqrtf(2.0f * n + 1.0f);
    float mv = mb[n * MSZ + mi];
    float Am = -rn * p - (float)(n + 1) * mv;     // (A m)[n], uses p = sum_{k<n} r_k m_k
    p = fmaf(rn, mv, p);
    float xv = mv + hdt * Am + dtu * rn;          // x[n]
    float y  = (xv - hdt * rn * q) / (1.0f + hdt * (float)(n + 1));  // fwd subst
    q = fmaf(rn, y, q);
    mb[n * MSZ + mi] = y;                         // in-place: (b,mi,n) read-once
  }
}

// ---------------- pack xm into WMMA bf16 A-operand layout ----------------
// 16-bit A 16x32 contiguous k order as B: lane<16 -> M=lane, K in {0..15};
// lane>=16 -> M=lane-16, K in {16..31}.  Rows (batches) 4..15 zeroed.
__global__ void k_pack_a(const float* __restrict__ m_t,
                         unsigned short* __restrict__ apack) {
  int tid   = blockIdx.x * blockDim.x + threadIdx.x;  // 0..65535
  int chunk = tid >> 5;
  int lane  = tid & 31;
  int k0    = chunk * 32;
  int row   = lane & 15;
  unsigned out[8];
  #pragma unroll
  for (int j = 0; j < 8; ++j) {
    unsigned short v2[2];
    #pragma unroll
    for (int e = 0; e < 2; ++e) {
      int i  = 2 * j + e;
      int kk = i + ((lane >= 16) ? 16 : 0);
      int K  = k0 + kk;                         // flat k = mi*NORD + n
      float v = 0.0f;
      if (row < BATCH) v = m_t[(size_t)row * MN + (size_t)(K & 255) * MSZ + (K >> 8)];
      v2[e] = f2bf(v);
    }
    out[j] = (unsigned)v2[0] | ((unsigned)v2[1] << 16);
  }
  unsigned* dst = (unsigned*)(apack + (size_t)tid * 16);  // 32B aligned
  *(uint4*)(dst)     = make_uint4(out[0], out[1], out[2], out[3]);
  *(uint4*)(dst + 4) = make_uint4(out[4], out[5], out[6], out[7]);
}

// ---------------- WMMA GEMM: xm @ W^T + bias, tanh / sigmoid ----------------
// blockIdx.x: [0..31] -> tanh path (W_hxm), [32..63] -> sigmoid path (W_gxm);
// 16-column h-tile per block, 8 waves split the K=65536 reduction, LDS reduce.
__global__ void __launch_bounds__(256)
k_gemm(const unsigned short* __restrict__ apack,
       const unsigned short* __restrict__ WhB,
       const unsigned short* __restrict__ WgB,
       const float* __restrict__ b_hxm, const float* __restrict__ b_gxm,
       float* __restrict__ hid, float* __restrict__ g) {
  int matsel = blockIdx.x >> 5;                 // 0=hid, 1=gate
  int h0     = (blockIdx.x & 31) * 16;
  int tid    = threadIdx.x;
  int wave   = tid >> 5;
  int lane   = tid & 31;

  const unsigned short* W = matsel ? WgB : WhB;
  // B-operand 16-bit 32x16: lanes 0-15 hold K=0..15 of column N=lane,
  // lanes 16-31 hold K=16..31 of column N=lane-16 -> 16 contiguous bf16 per lane.
  int col = h0 + (lane & 15);
  const unsigned short* wrow = W + (size_t)col * MN + ((lane < 16) ? 0 : 16);

  v8f c = {};
  int c0 = wave * (KCHUNKS / 8), c1 = c0 + (KCHUNKS / 8);
  for (int ch = c0; ch < c1; ++ch) {
    int k0 = ch * 32;
    __builtin_prefetch(wrow + k0 + 4096, 0, 1);           // global_prefetch_b8
    v16bf a  = *(const v16bf*)(apack + (size_t)ch * 512 + (size_t)lane * 16);
    v16bf bm = *(const v16bf*)(wrow + k0);
    c = __builtin_amdgcn_wmma_f32_16x16x32_bf16(false, a, false, bm,
                                                (short)0, c, false, false);
  }

  __shared__ float red[8][32][8];
  #pragma unroll
  for (int j = 0; j < 8; ++j) red[wave][lane][j] = c[j];
  __syncthreads();

  // C layout: lanes 0-15 => N=lane, VGPR j => M=j.  Real batches are M=0..3.
  if (wave == 0 && lane < 16) {
    int hh = h0 + lane;
    float bias = matsel ? b_gxm[hh] : b_hxm[hh];
    #pragma unroll
    for (int j = 0; j < BATCH; ++j) {
      float s = 0.0f;
      #pragma unroll
      for (int w = 0; w < 8; ++w) s += red[w][lane][j];
      s += bias;
      if (matsel) g  [j * HID + hh] = 1.0f / (1.0f + __expf(-s));
      else        hid[j * HID + hh] = tanhf(s);
    }
  }
}

// ---------------- gated hidden update ----------------
__global__ void k_hupdate(float* __restrict__ h, const float* __restrict__ hid,
                          const float* __restrict__ g) {
  int i = blockIdx.x * blockDim.x + threadIdx.x;
  if (i < BATCH * HID) h[i] = (1.0f - g[i]) * h[i] + g[i] * hid[i];
}

__global__ void k_copy(const float* __restrict__ h, float* __restrict__ out) {
  int i = blockIdx.x * blockDim.x + threadIdx.x;
  if (i < BATCH * HID) out[i] = h[i];
}

// ---------------- host launcher ----------------
extern "C" void kernel_launch(void* const* d_in, const int* in_sizes, int n_in,
                              void* d_out, int out_size, void* d_ws, size_t ws_size,
                              hipStream_t stream) {
  const float* ts    = (const float*)d_in[0];
  const float* W_uxh = (const float*)d_in[1];
  const float* b_uxh = (const float*)d_in[2];
  const float* W_hxm = (const float*)d_in[3];
  const float* b_hxm = (const float*)d_in[4];
  const float* W_gxm = (const float*)d_in[5];
  const float* b_gxm = (const float*)d_in[6];

  char* ws = (char*)d_ws;
  float*          h     = (float*)(ws + OFF_H);
  float*          hid   = (float*)(ws + OFF_HID);
  float*          g     = (float*)(ws + OFF_G);
  float*          m_t   = (float*)(ws + OFF_M);
  unsigned short* apack = (unsigned short*)(ws + OFF_APACK);
  unsigned short* WhB   = (unsigned short*)(ws + OFF_WH);
  unsigned short* WgB   = (unsigned short*)(ws + OFF_WG);

  k_zero<<<(ZERO_FLOATS + 255) / 256, 256, 0, stream>>>((float*)ws);
  k_cvt<<<4096, 256, 0, stream>>>(W_hxm, W_gxm, WhB, WgB);

  for (int t = 0; t < SEQLEN; ++t) {
    k_memstep<<<BATCH, MSZ, 0, stream>>>(ts, W_uxh, b_uxh, h, m_t, t);
    k_pack_a<<<256, 256, 0, stream>>>(m_t, apack);
    k_gemm<<<64, 256, 0, stream>>>(apack, WhB, WgB, b_hxm, b_gxm, hid, g);
    k_hupdate<<<8, 256, 0, stream>>>(h, hid, g);
  }
  k_copy<<<8, 256, 0, stream>>>(h, (float*)d_out);
}
